// GraphEncoder_54554674593853
// MI455X (gfx1250) — matrix operand order repaired and checked by reading, not compiled, not run on hardware
//
#include <hip/hip_runtime.h>

#define NEG_SLOPE 0.2f
#define NL 4
#define NN 10000
#define NE 40000
#define NR 2000
#define DD 128
#define HH 8
#define HD 1024  // H * D

#define USE_ASYNC_LDS 1

typedef __attribute__((ext_vector_type(16))) __bf16          v16bf;
typedef __attribute__((ext_vector_type(8)))  float           v8f;
typedef __attribute__((ext_vector_type(4)))  unsigned int    u32x4;
typedef __attribute__((ext_vector_type(4)))  float           f32x4;
typedef __attribute__((ext_vector_type(4)))  unsigned short  u16x4;

union FragBF { v16bf v; u32x4 q[2]; };

__device__ __forceinline__ unsigned short f2bf(float f) {
  unsigned u = __float_as_uint(f);
  u += 0x7FFFu + ((u >> 16) & 1u);   // round-to-nearest-even to bf16
  return (unsigned short)(u >> 16);
}

__device__ __forceinline__ void atomicMaxF(float* a, float v) {
  if (v >= 0.f) atomicMax((int*)a, __float_as_int(v));
  else          atomicMin((unsigned int*)a, __float_as_uint(v));
}

// ---------------------------------------------------------------------------
// C[M,N] = A_bf16[M,K] @ WT_bf16[N,K]^T + bias[N]
// Block: 256 thr = 8 waves; tile 16(M) x 128(N); wave tile 16x16; K panels 128.
// All fragment reads are contiguous, swizzled 16B chunks -> ds_load_b128.
// B panel staged via global_load_async_to_lds_b128 (ASYNCcnt path).
// All addressing in 32-bit (buffers << 2^31 elems) to enable strength reduction.
// ---------------------------------------------------------------------------
__global__ __launch_bounds__(256)
void gemm_wmma(const unsigned short* __restrict__ A,
               const unsigned short* __restrict__ WT,
               const float* __restrict__ bias, float* __restrict__ C,
               int M, int K, int N)
{
  __shared__ __align__(16) unsigned short As[16 * 128];
  __shared__ __align__(16) unsigned short Bs[128 * 128];
  const int tid  = threadIdx.x;
  const int lane = tid & 31;
  const int wv   = tid >> 5;
  const unsigned m0 = blockIdx.y * 16;
  const unsigned n0 = blockIdx.x * 128;
  const int half = lane >> 4;
  const unsigned uK = (unsigned)K, uN = (unsigned)N;

  v8f acc = {};
  for (unsigned kp = 0; kp < uK; kp += 128) {
    { // A tile: 16 rows x 128 k = 256 chunks of 8 shorts, swizzled (blk ^ row)
      unsigned r = tid >> 4, c = tid & 15;
      const u32x4* g = (const u32x4*)(A + (m0 + r) * uK + kp + c * 8);
      *(u32x4*)(As + r * 128 + (((c ^ r) & 15) << 3)) = *g;
    }
    { // B panel: 128 n-rows x 128 k, 16-short blocks swizzled by (blk ^ (n&7))
      unsigned n = tid >> 4, c = tid & 15;
      const unsigned short* g = WT + (n0 + n) * uK + kp + c * 8;
      unsigned loff0 = (unsigned)(uintptr_t)(Bs + n * 128 +
                       ((((c >> 1) ^ (n & 7)) << 4) | ((c & 1) << 3)));
      #pragma unroll
      for (int it = 0; it < 8; ++it) {   // n stride 16 keeps (n&7) invariant
#if USE_ASYNC_LDS
        asm volatile("global_load_async_to_lds_b128 %0, %1, off"
                     :: "v"(loff0 + (unsigned)it * 4096u),
                        "v"((unsigned long long)(uintptr_t)g)
                     : "memory");
#else
        *(u32x4*)(uintptr_t)(loff0 + (unsigned)it * 4096u) = *(const u32x4*)g;
#endif
        g += (uK << 4);                  // advance 16 n-rows
      }
    }
    if (kp + 128 < uK)
      __builtin_prefetch(WT + (n0 + (tid >> 1)) * uK + kp + 128 + (tid & 1) * 64, 0, 1);
#if USE_ASYNC_LDS
    asm volatile("s_wait_asynccnt 0" ::: "memory");
#endif
    __syncthreads();

    #pragma unroll
    for (int kk = 0; kk < 128; kk += 32) {
      const int mrow = lane & 15;
      const int nn   = wv * 16 + (lane & 15);
      FragBF a, b;
      int ba0 = (kk >> 3) + 2 * half;             // A: k = kk + 16*half, 8-short block
      a.q[0] = *(const u32x4*)(As + mrow * 128 + ((( ba0      ^ mrow) & 15) << 3));
      a.q[1] = *(const u32x4*)(As + mrow * 128 + ((((ba0 + 1) ^ mrow) & 15) << 3));
      int bb0 = ((((kk >> 4) + half) ^ (nn & 7)) << 4);  // B: k = kk + 16*half
      b.q[0] = *(const u32x4*)(Bs + nn * 128 + bb0);
      b.q[1] = *(const u32x4*)(Bs + nn * 128 + bb0 + 8);
      acc = __builtin_amdgcn_wmma_f32_16x16x32_bf16(false, a.v, false, b.v,
                                                    (short)0, acc, false, false);
    }
    __syncthreads();
  }

  const int col = lane & 15;
  const unsigned gn = n0 + wv * 16 + col;
  const float bv = bias ? bias[gn] : 0.f;
  #pragma unroll
  for (int r8 = 0; r8 < 8; ++r8)
    C[(m0 + r8 + 8 * half) * uN + gn] = acc[r8] + bv;
}

// ---------------------------------------------------------------------------
// out[n*K + k] = bf16(in[k*N + n])  -- weight transpose+convert (once/launch)
__global__ __launch_bounds__(256)
void transpose_bf16_kernel(const float* __restrict__ in,
                           unsigned short* __restrict__ out, int K, int N)
{
  __shared__ unsigned short tile[64][65];
  unsigned k0 = blockIdx.y * 64, n0 = blockIdx.x * 64;
  for (int i = threadIdx.x; i < 4096; i += 256) {
    unsigned kr = i >> 6, nc = i & 63;
    tile[kr][nc] = f2bf(in[(k0 + kr) * (unsigned)N + n0 + nc]);
  }
  __syncthreads();
  for (int i = threadIdx.x; i < 4096; i += 256) {
    unsigned nr = i >> 6, kc = i & 63;
    out[(n0 + nr) * (unsigned)K + k0 + kc] = tile[kc][nr];
  }
}

// elementwise f32 -> bf16 (optionally relu), 4 elements/thread
__global__ void convert_bf16_kernel(const float* __restrict__ in,
                                    unsigned short* __restrict__ out, int n4, int relu)
{
  int t = blockIdx.x * blockDim.x + threadIdx.x;
  if (t >= n4) return;
  f32x4 v = ((const f32x4*)in)[t];
  if (relu) {
    v[0] = fmaxf(v[0], 0.f); v[1] = fmaxf(v[1], 0.f);
    v[2] = fmaxf(v[2], 0.f); v[3] = fmaxf(v[3], 0.f);
  }
  u16x4 o;
  o[0] = f2bf(v[0]); o[1] = f2bf(v[1]); o[2] = f2bf(v[2]); o[3] = f2bf(v[3]);
  ((u16x4*)out)[t] = o;
}

// ---------------------------------------------------------------------------
__global__ void fill_kernel(float* p, int n, float v) {
  int t = blockIdx.x * blockDim.x + threadIdx.x;
  if (t < n) p[t] = v;
}

__global__ __launch_bounds__(128)
void scatter_loop_kernel(const float* __restrict__ rel, const int* __restrict__ ridx,
                         const int* __restrict__ dst, float* __restrict__ lsum,
                         float* __restrict__ deg)
{
  unsigned e = blockIdx.x, c = threadIdx.x;
  unsigned d = (unsigned)dst[e];
  atomicAdd(&lsum[d * DD + c], rel[(unsigned)ridx[e] * DD + c]);
  if (c == 0) atomicAdd(&deg[d], 1.f);
}

__global__ void loopdiv_kernel(float* lsum, const float* deg) {
  int t = blockIdx.x * blockDim.x + threadIdx.x;
  if (t >= NN * DD) return;
  lsum[t] = lsum[t] / fmaxf(deg[t >> 7], 1.f);
}

// alpha = leakyrelu(xl[src]+xr[dst]+eproj) . att   (1 wave32 per (edge,head))
__global__ __launch_bounds__(256)
void alpha1_kernel(const float* __restrict__ XL, const float* __restrict__ XR,
                   const float* __restrict__ RE, const float* __restrict__ LP,
                   const int* __restrict__ src, const int* __restrict__ dst,
                   const int* __restrict__ ridx, const float* __restrict__ att,
                   float* __restrict__ araw, float* __restrict__ nmax)
{
  unsigned e = blockIdx.x;
  unsigned h = threadIdx.x >> 5;
  unsigned lane = threadIdx.x & 31;
  unsigned s, d; const float* ep;
  if (e < NE) { s = (unsigned)src[e]; d = (unsigned)dst[e]; ep = RE + (unsigned)ridx[e] * HD; }
  else        { s = d = e - NE;                             ep = LP + (e - NE) * HD; }
  const float* xl = XL + s * HD + h * DD;
  const float* xr = XR + d * HD + h * DD;
  const float* pp = ep + h * DD;
  const float* at = att + h * DD;
  float acc = 0.f;
  for (unsigned c = lane; c < DD; c += 32) {
    float m = xl[c] + xr[c] + pp[c];
    m = m > 0.f ? m : NEG_SLOPE * m;
    acc += m * at[c];
  }
  for (int off = 16; off; off >>= 1) acc += __shfl_down(acc, off, 32);
  if (lane == 0) {
    araw[e * HH + h] = acc;
    atomicMaxF(&nmax[d * HH + h], acc);
  }
}

__global__ void softmax2_kernel(const float* __restrict__ araw, const int* __restrict__ dst,
                                const float* __restrict__ nmax, float* __restrict__ wexp,
                                float* __restrict__ denom)
{
  unsigned t = blockIdx.x * blockDim.x + threadIdx.x;
  if (t >= (NE + NN) * HH) return;
  unsigned e = t / HH, h = t - e * HH;
  unsigned d = (e < NE) ? (unsigned)dst[e] : (e - NE);
  float w = __expf(araw[t] - nmax[d * HH + h]);
  wexp[t] = w;
  atomicAdd(&denom[d * HH + h], w);
}

// fused aggregation + head sum: xacc[dst,c] += sum_h alpha * XL[src,h,c]
__global__ __launch_bounds__(128)
void aggregate_kernel(const float* __restrict__ XL, const float* __restrict__ wexp,
                      const float* __restrict__ denom, const int* __restrict__ src,
                      const int* __restrict__ dst, float* __restrict__ xacc)
{
  unsigned e = blockIdx.x, c = threadIdx.x;
  unsigned s, d;
  if (e < NE) { s = (unsigned)src[e]; d = (unsigned)dst[e]; } else { s = d = e - NE; }
  const float* xl = XL + s * HD;
  float acc = 0.f;
  #pragma unroll
  for (int h = 0; h < HH; ++h)
    acc += (wexp[e * HH + h] / denom[d * HH + h]) * xl[h * DD + c];
  atomicAdd(&xacc[d * DD + c], acc);
}

__global__ void finalize_x_kernel(float* x, const float* __restrict__ bias) {
  int t = blockIdx.x * blockDim.x + threadIdx.x;
  if (t >= NN * DD) return;
  x[t] = x[t] * (1.f / HH) + bias[t & (DD - 1)];
}

// ---------------------------------------------------------------------------
extern "C" void kernel_launch(void* const* d_in, const int* in_sizes, int n_in,
                              void* d_out, int out_size, void* d_ws, size_t ws_size,
                              hipStream_t stream)
{
  (void)in_sizes; (void)n_in; (void)out_size; (void)ws_size;
  const float* x_in   = (const float*)d_in[0];
  const int*   eidx   = (const int*)d_in[1];
  const float* rel_in = (const float*)d_in[2];
  const int*   ridx   = (const int*)d_in[3];
  const float* Wl = (const float*)d_in[4];
  const float* bl = (const float*)d_in[5];
  const float* Wr = (const float*)d_in[6];
  const float* br = (const float*)d_in[7];
  const float* We = (const float*)d_in[8];
  const float* att= (const float*)d_in[9];
  const float* bi = (const float*)d_in[10];
  const float* Wb = (const float*)d_in[11];
  const float* bb = (const float*)d_in[12];
  const int* src = eidx;
  const int* dst = eidx + NE;

  float* f = (float*)d_ws;
  float* xA   = f; f += (size_t)NN * DD;
  float* xB   = f; f += (size_t)NN * DD;
  float* rA   = f; f += (size_t)NR * DD;
  float* rB   = f; f += (size_t)NR * DD;
  float* XL   = f; f += (size_t)NN * HD;
  float* XR   = f; f += (size_t)NN * HD;
  float* LP   = f; f += (size_t)NN * HD;
  float* RE   = f; f += (size_t)NR * HD;
  float* lsum = f; f += (size_t)NN * DD;
  float* deg  = f; f += NN;
  float* araw = f; f += (size_t)(NE + NN) * HH;
  float* wexp = f; f += (size_t)(NE + NN) * HH;
  float* nmax = f; f += (size_t)NN * HH;
  float* denm = f; f += (size_t)NN * HH;

  unsigned short* u = (unsigned short*)f;
  unsigned short* WlT = u; u += (size_t)NL * HD * DD;   // [l][1024][128]
  unsigned short* WrT = u; u += (size_t)NL * HD * DD;
  unsigned short* WeT = u; u += (size_t)NL * HD * DD;
  unsigned short* WbT = u; u += (size_t)NL * DD * HD;   // [l][128][1024]
  unsigned short* xbf = u; u += (size_t)NN * DD;
  unsigned short* lsbf= u; u += (size_t)NN * DD;
  unsigned short* rbf = u; u += (size_t)NR * DD;
  unsigned short* REbf= u; u += (size_t)NR * HD;

  // weights -> transposed bf16 (every launch; deterministic)
  for (int l = 0; l < NL; ++l) {
    dim3 gW(HD / 64, DD / 64);   // K=128, N=1024
    transpose_bf16_kernel<<<gW, 256, 0, stream>>>(Wl + (size_t)l * DD * HD,
                                                  WlT + (size_t)l * HD * DD, DD, HD);
    transpose_bf16_kernel<<<gW, 256, 0, stream>>>(Wr + (size_t)l * DD * HD,
                                                  WrT + (size_t)l * HD * DD, DD, HD);
    transpose_bf16_kernel<<<gW, 256, 0, stream>>>(We + (size_t)l * DD * HD,
                                                  WeT + (size_t)l * HD * DD, DD, HD);
    dim3 gB(DD / 64, HD / 64);   // K=1024, N=128
    transpose_bf16_kernel<<<gB, 256, 0, stream>>>(Wb + (size_t)l * HD * DD,
                                                  WbT + (size_t)l * DD * HD, HD, DD);
  }

  hipMemcpyAsync(xA, x_in, (size_t)NN * DD * 4, hipMemcpyDeviceToDevice, stream);
  hipMemcpyAsync(rA, rel_in, (size_t)NR * DD * 4, hipMemcpyDeviceToDevice, stream);

  float* xc = xA; float* xn = xB; float* rc = rA; float* rn = rB;

  for (int l = 0; l < NL; ++l) {
    const float* bl_l = bl + (size_t)l * HD;
    const float* br_l = br + (size_t)l * HD;
    const float* at_l = att + (size_t)l * HH * DD;
    const float* bi_l = bi + (size_t)l * DD;
    const float* bb_l = bb + (size_t)l * DD;
    const unsigned short* WlT_l = WlT + (size_t)l * HD * DD;
    const unsigned short* WrT_l = WrT + (size_t)l * HD * DD;
    const unsigned short* WeT_l = WeT + (size_t)l * HD * DD;
    const unsigned short* WbT_l = WbT + (size_t)l * DD * HD;

    fill_kernel<<<(NN * DD + 255) / 256, 256, 0, stream>>>(lsum, NN * DD, 0.f);
    fill_kernel<<<(NN + 255) / 256, 256, 0, stream>>>(deg, NN, 0.f);
    fill_kernel<<<(NN * HH + 255) / 256, 256, 0, stream>>>(nmax, NN * HH, -__builtin_inff());
    fill_kernel<<<(NN * HH + 255) / 256, 256, 0, stream>>>(denm, NN * HH, 0.f);
    fill_kernel<<<(NN * DD + 255) / 256, 256, 0, stream>>>(xn, NN * DD, 0.f);

    scatter_loop_kernel<<<NE, 128, 0, stream>>>(rc, ridx, dst, lsum, deg);
    loopdiv_kernel<<<(NN * DD + 255) / 256, 256, 0, stream>>>(lsum, deg);

    convert_bf16_kernel<<<(NN * DD / 4 + 255) / 256, 256, 0, stream>>>(xc, xbf, NN * DD / 4, 0);
    convert_bf16_kernel<<<(NN * DD / 4 + 255) / 256, 256, 0, stream>>>(lsum, lsbf, NN * DD / 4, 0);
    convert_bf16_kernel<<<(NR * DD / 4 + 255) / 256, 256, 0, stream>>>(rc, rbf, NR * DD / 4, 0);

    dim3 gN(HD / 128, NN / 16);
    gemm_wmma<<<gN, 256, 0, stream>>>(xbf,  WlT_l, bl_l,    XL, NN, DD, HD);
    gemm_wmma<<<gN, 256, 0, stream>>>(xbf,  WrT_l, br_l,    XR, NN, DD, HD);
    gemm_wmma<<<gN, 256, 0, stream>>>(lsbf, WeT_l, nullptr, LP, NN, DD, HD);
    dim3 gR(HD / 128, NR / 16);
    gemm_wmma<<<gR, 256, 0, stream>>>(rbf,  WeT_l, nullptr, RE, NR, DD, HD);

    alpha1_kernel<<<NE + NN, 256, 0, stream>>>(XL, XR, RE, LP, src, dst, ridx, at_l, araw, nmax);
    softmax2_kernel<<<((NE + NN) * HH + 255) / 256, 256, 0, stream>>>(araw, dst, nmax, wexp, denm);
    aggregate_kernel<<<NE + NN, 128, 0, stream>>>(XL, wexp, denm, src, dst, xn);
    finalize_x_kernel<<<(NN * DD + 255) / 256, 256, 0, stream>>>(xn, bi_l);

    convert_bf16_kernel<<<(NR * HD / 4 + 255) / 256, 256, 0, stream>>>(RE, REbf, NR * HD / 4, 1);
    dim3 gU(DD / 128, NR / 16);
    gemm_wmma<<<gU, 256, 0, stream>>>(REbf, WbT_l, bb_l, rn, NR, HD, DD);

    float* t1 = xc; xc = xn; xn = t1;
    float* t2 = rc; rc = rn; rn = t2;
  }

  hipMemcpyAsync(d_out, xc, (size_t)NN * DD * 4, hipMemcpyDeviceToDevice, stream);
  hipMemcpyAsync((float*)d_out + (size_t)NN * DD, rc, (size_t)NR * DD * 4,
                 hipMemcpyDeviceToDevice, stream);
}
